// SparseMLP_10033043604250
// MI455X (gfx1250) — matrix operand order repaired and checked by reading, not compiled, not run on hardware
//
#include <hip/hip_runtime.h>
#include <hip/hip_bf16.h>
#include <stdint.h>

// ---------------- problem constants ----------------
#define INPUT_DIM   1024
#define HIDDEN_DIM  4096
#define OUTPUT_DIM  1024
#define SPARSITY    1024
#define BROWS       16384            // 8 * 2048
#define ALPHA_C     5.0f
#define BETA_C      0.5f

// ---------------- tiling ----------------
#define BM   128
#define BN   128
#define BK   32
#define LDSS 40                      // padded LDS row stride in bf16 (80B = 5x16B, conflict-spread)

typedef __attribute__((ext_vector_type(16))) __bf16 v16bf;
typedef __attribute__((ext_vector_type(8)))  float  v8f;

union FragBF16 { v16bf v; uint4 u[2]; };

// ---- f32 -> bf16 RNE (v_perm_b32-packed; HW cvt_pk not available on gfx1250) ----
__device__ __forceinline__ unsigned int pack2_bf16(float a, float b) {
  unsigned int ua = __float_as_uint(a);
  unsigned int ub = __float_as_uint(b);
  ua += 0x7FFFu + ((ua >> 16) & 1u);
  ub += 0x7FFFu + ((ub >> 16) & 1u);
  return __builtin_amdgcn_perm(ub, ua, 0x07060302u);   // {ub.hi16, ua.hi16}
}
__device__ __forceinline__ unsigned short f32_to_bf16_bits(float f) {
  unsigned int u = __float_as_uint(f);
  u += 0x7FFFu + ((u >> 16) & 1u);
  return (unsigned short)(u >> 16);
}

// ---- CDNA5 async global->LDS copy (ASYNCcnt path, CDNA5 ISA) ----
__device__ __forceinline__ void async_ld_b128(unsigned lds_addr, const void* gaddr) {
  asm volatile("global_load_async_to_lds_b128 %0, %1, off"
               :: "v"(lds_addr), "v"(gaddr) : "memory");
}
__device__ __forceinline__ unsigned lds_off(const void* p) {
  return (unsigned)(size_t)p;        // low 32 bits of flat addr = LDS offset
}

// =====================================================================
// Kernel C1: streaming f32 -> bf16 (x and W_in)
// =====================================================================
__global__ __launch_bounds__(256)
void convert_f32_bf16(const float* __restrict__ src,
                      unsigned short* __restrict__ dst, int n4)
{
  const int i = blockIdx.x * blockDim.x + threadIdx.x;
  if (i < n4) {
    const float4 v = ((const float4*)src)[i];
    ((uint2*)dst)[i] = make_uint2(pack2_bf16(v.x, v.y), pack2_bf16(v.z, v.w));
  }
}

// =====================================================================
// Kernel C2: W_out[:SPARSITY] f32 -> bf16 with transpose (LDS-tiled 64x64)
// WoutT[o][s] = bf16(W_out[s][o]); makes GEMM2's B tile a contiguous copy.
// =====================================================================
__global__ __launch_bounds__(256)
void transpose_convert_wout(const float* __restrict__ Wout,
                            unsigned short* __restrict__ WoutT)
{
  __shared__ unsigned short t_lds[64][65];       // +1 pad: bank-spread
  const int tid = threadIdx.x;
  const int s0  = blockIdx.x * 64;
  const int o0  = blockIdx.y * 64;
#pragma unroll
  for (int i = 0; i < 4; ++i) {
    const int idx = tid + i * 256;
    const int r = idx >> 4;                      // s within tile
    const int c = (idx & 15) * 4;                // o within tile
    const float4 v = *(const float4*)(Wout + (size_t)(s0 + r) * OUTPUT_DIM + o0 + c);
    t_lds[c + 0][r] = f32_to_bf16_bits(v.x);
    t_lds[c + 1][r] = f32_to_bf16_bits(v.y);
    t_lds[c + 2][r] = f32_to_bf16_bits(v.z);
    t_lds[c + 3][r] = f32_to_bf16_bits(v.w);
  }
  __syncthreads();
#pragma unroll
  for (int i = 0; i < 4; ++i) {
    const int idx = tid + i * 256;
    const int orow = idx >> 4;                   // o within tile
    const int sc   = (idx & 15) * 4;             // s within tile
    const unsigned int u0 = (unsigned int)t_lds[orow][sc + 0] |
                            ((unsigned int)t_lds[orow][sc + 1] << 16);
    const unsigned int u1 = (unsigned int)t_lds[orow][sc + 2] |
                            ((unsigned int)t_lds[orow][sc + 3] << 16);
    *(uint2*)(WoutT + (size_t)(o0 + orow) * SPARSITY + s0 + sc) = make_uint2(u0, u1);
  }
}

// =====================================================================
// Kernel 1: z = x @ W_in^T  (bf16 WMMA, f32 acc), async double-buffered
// LDS tiles; fused clip+sigmoid; writes top' = p/(p+1e-8) (bf16) and
// per-(row, n-block) partial row sums (deterministic order).
// grid = (BROWS/BM, HIDDEN_DIM/BN) = (128, 32), block = 256 (8 wave32)
// =====================================================================
__global__ __launch_bounds__(256)
void sparse_mlp_gemm1_act(const unsigned short* __restrict__ xb,
                          const unsigned short* __restrict__ Winb,
                          unsigned short* __restrict__ topc,
                          float* __restrict__ partial)
{
  __shared__ unsigned short a_lds[2][BM * LDSS];
  __shared__ unsigned short b_lds[2][BN * LDSS];
  __shared__ float rs_lds[BM * 4];

  const int tid  = threadIdx.x;
  const int wave = tid >> 5;
  const int lane = tid & 31;
  const int wm   = (wave >> 2) * 64;
  const int wcol = wave & 3;
  const int wn   = wcol * 32;
  const int m0   = blockIdx.x * BM;
  const int n0   = blockIdx.y * BN;
  const int nl   = lane & 15;
  const int hs   = lane >> 4;

  v8f acc[4][2];
  const v8f vzero = {0.f, 0.f, 0.f, 0.f, 0.f, 0.f, 0.f, 0.f};
#pragma unroll
  for (int mi = 0; mi < 4; ++mi)
#pragma unroll
    for (int ni = 0; ni < 2; ++ni) acc[mi][ni] = vzero;

  // 4 async B128 instructions per wave per tile-pair (A:2 + B:2 per thread-slice)
  auto issue_tile = [&](int buf, int k0) {
#pragma unroll
    for (int i = 0; i < 2; ++i) {
      const int idx = tid + i * 256;
      const int row = idx >> 2;                  // 0..127
      const int kq  = (idx & 3) * 8;             // 0,8,16,24 (bf16 elems)
      async_ld_b128(lds_off(&a_lds[buf][row * LDSS + kq]),
                    xb   + (size_t)(m0 + row) * INPUT_DIM + k0 + kq);
      async_ld_b128(lds_off(&b_lds[buf][row * LDSS + kq]),
                    Winb + (size_t)(n0 + row) * INPUT_DIM + k0 + kq);
    }
  };

  issue_tile(0, 0);
  int buf = 0;
  for (int k0 = 0; k0 < INPUT_DIM; k0 += BK, buf ^= 1) {
    if (k0 + BK < INPUT_DIM) {
      issue_tile(buf ^ 1, k0 + BK);              // next tile in flight
      asm volatile("s_wait_asynccnt 0x4" ::: "memory");   // current tile landed
    } else {
      asm volatile("s_wait_asynccnt 0x0" ::: "memory");
    }
    __syncthreads();                             // all waves' copies visible

    FragBF16 bfrag[2];
#pragma unroll
    for (int ni = 0; ni < 2; ++ni) {
      const int n    = wn + ni * 16 + nl;
      const int koff = hs * 16;
      bfrag[ni].u[0] = *(const uint4*)(&b_lds[buf][n * LDSS + koff]);
      bfrag[ni].u[1] = *(const uint4*)(&b_lds[buf][n * LDSS + koff + 8]);
    }
#pragma unroll
    for (int mi = 0; mi < 4; ++mi) {
      FragBF16 afrag;
      const int m  = wm + mi * 16 + nl;
      const int kb = hs * 16;
      afrag.u[0] = *(const uint4*)(&a_lds[buf][m * LDSS + kb]);
      afrag.u[1] = *(const uint4*)(&a_lds[buf][m * LDSS + kb + 8]);
#pragma unroll
      for (int ni = 0; ni < 2; ++ni) {
        acc[mi][ni] = __builtin_amdgcn_wmma_f32_16x16x32_bf16(
            false, afrag.v, false, bfrag[ni].v, (short)0, acc[mi][ni], false, false);
      }
    }
    __syncthreads();                             // reads done: buf re-issuable
  }

  // ---- fused epilogue: clip + sigmoid; deterministic row-sum reduction ----
#pragma unroll
  for (int mi = 0; mi < 4; ++mi) {
#pragma unroll
    for (int ni = 0; ni < 2; ++ni)
#pragma unroll
      for (int r = 0; r < 8; ++r) {
        float z = acc[mi][ni][r];
        z = fminf(10.0f, fmaxf(-10.0f, z));
        acc[mi][ni][r] = 1.0f / (1.0f + __expf(-ALPHA_C * (z - BETA_C)));
      }
#pragma unroll
    for (int r = 0; r < 8; ++r) {
      float s = acc[mi][0][r] + acc[mi][1][r];
      s += __shfl_xor(s, 1, 32);
      s += __shfl_xor(s, 2, 32);
      s += __shfl_xor(s, 4, 32);
      s += __shfl_xor(s, 8, 32);                 // within 16-lane half
      if (nl == 0) rs_lds[(wm + mi * 16 + hs * 8 + r) * 4 + wcol] = s;
    }
  }

  // ---- write top' = p/(p+1e-8) for the first SPARSITY hidden units ----
  if (n0 < SPARSITY) {
#pragma unroll
    for (int mi = 0; mi < 4; ++mi)
#pragma unroll
      for (int ni = 0; ni < 2; ++ni)
#pragma unroll
        for (int r = 0; r < 8; ++r) {
          const int M = m0 + wm + mi * 16 + hs * 8 + r;
          const int N = n0 + wn + ni * 16 + nl;
          const float p = acc[mi][ni][r];
          topc[(size_t)M * SPARSITY + N] = f32_to_bf16_bits(p / (p + 1e-8f));
        }
  }

  __syncthreads();
  if (tid < BM) {   // 4 wave-columns combined in fixed order -> deterministic
    const float t = rs_lds[tid * 4 + 0] + rs_lds[tid * 4 + 1] +
                    rs_lds[tid * 4 + 2] + rs_lds[tid * 4 + 3];
    partial[(size_t)(m0 + tid) * gridDim.y + blockIdx.y] = t;
  }
}

// =====================================================================
// Kernel 2: norm[row] = (sum over n-blocks of partial) / SPARSITY
// =====================================================================
__global__ __launch_bounds__(256)
void sparse_mlp_rownorm(const float* __restrict__ partial,
                        float* __restrict__ norm, int nblocks)
{
  const int row = blockIdx.x * blockDim.x + threadIdx.x;
  if (row < BROWS) {
    float s = 0.0f;
    const float4* p4 = (const float4*)(partial + (size_t)row * nblocks);
    for (int i = 0; i < nblocks / 4; ++i) {
      const float4 v = p4[i];
      s += v.x + v.y + v.z + v.w;
    }
    norm[row] = s * (1.0f / (float)SPARSITY);
  }
}

// =====================================================================
// Kernel 3: out = norm[b] * (top' @ WoutT^T); pure-bf16 async-copy GEMM,
// per-row norm scale applied in the f32 epilogue.
// grid = (BROWS/BM, OUTPUT_DIM/BN) = (128, 8)
// =====================================================================
__global__ __launch_bounds__(256)
void sparse_mlp_gemm2(const unsigned short* __restrict__ topc,
                      const float* __restrict__ norm,
                      const unsigned short* __restrict__ WoutT,
                      float* __restrict__ out)
{
  __shared__ unsigned short a_lds[2][BM * LDSS];
  __shared__ unsigned short b_lds[2][BN * LDSS];
  __shared__ float nrm_lds[BM];

  const int tid  = threadIdx.x;
  const int wave = tid >> 5;
  const int lane = tid & 31;
  const int wm   = (wave >> 2) * 64;
  const int wn   = (wave & 3) * 32;
  const int m0   = blockIdx.x * BM;
  const int n0   = blockIdx.y * BN;
  const int nl   = lane & 15;
  const int hs   = lane >> 4;

  if (tid < BM) nrm_lds[tid] = norm[m0 + tid];   // covered by in-loop barriers

  v8f acc[4][2];
  const v8f vzero = {0.f, 0.f, 0.f, 0.f, 0.f, 0.f, 0.f, 0.f};
#pragma unroll
  for (int mi = 0; mi < 4; ++mi)
#pragma unroll
    for (int ni = 0; ni < 2; ++ni) acc[mi][ni] = vzero;

  auto issue_tile = [&](int buf, int k0) {
#pragma unroll
    for (int i = 0; i < 2; ++i) {
      const int idx = tid + i * 256;
      const int row = idx >> 2;
      const int kq  = (idx & 3) * 8;
      async_ld_b128(lds_off(&a_lds[buf][row * LDSS + kq]),
                    topc  + (size_t)(m0 + row) * SPARSITY + k0 + kq);
      async_ld_b128(lds_off(&b_lds[buf][row * LDSS + kq]),
                    WoutT + (size_t)(n0 + row) * SPARSITY + k0 + kq);
    }
  };

  issue_tile(0, 0);
  int buf = 0;
  for (int k0 = 0; k0 < SPARSITY; k0 += BK, buf ^= 1) {
    if (k0 + BK < SPARSITY) {
      issue_tile(buf ^ 1, k0 + BK);
      asm volatile("s_wait_asynccnt 0x4" ::: "memory");
    } else {
      asm volatile("s_wait_asynccnt 0x0" ::: "memory");
    }
    __syncthreads();

    FragBF16 bfrag[2];
#pragma unroll
    for (int ni = 0; ni < 2; ++ni) {
      const int n    = wn + ni * 16 + nl;
      const int koff = hs * 16;
      bfrag[ni].u[0] = *(const uint4*)(&b_lds[buf][n * LDSS + koff]);
      bfrag[ni].u[1] = *(const uint4*)(&b_lds[buf][n * LDSS + koff + 8]);
    }
#pragma unroll
    for (int mi = 0; mi < 4; ++mi) {
      FragBF16 afrag;
      const int m  = wm + mi * 16 + nl;
      const int kb = hs * 16;
      afrag.u[0] = *(const uint4*)(&a_lds[buf][m * LDSS + kb]);
      afrag.u[1] = *(const uint4*)(&a_lds[buf][m * LDSS + kb + 8]);
#pragma unroll
      for (int ni = 0; ni < 2; ++ni) {
        acc[mi][ni] = __builtin_amdgcn_wmma_f32_16x16x32_bf16(
            false, afrag.v, false, bfrag[ni].v, (short)0, acc[mi][ni], false, false);
      }
    }
    __syncthreads();
  }

  // ---- f32 epilogue: scale by per-row norm, store ----
#pragma unroll
  for (int mi = 0; mi < 4; ++mi)
#pragma unroll
    for (int ni = 0; ni < 2; ++ni)
#pragma unroll
      for (int r = 0; r < 8; ++r) {
        const int lm = wm + mi * 16 + hs * 8 + r;
        const int N  = n0 + wn + ni * 16 + nl;
        out[(size_t)(m0 + lm) * OUTPUT_DIM + N] = acc[mi][ni][r] * nrm_lds[lm];
      }
}

// =====================================================================
extern "C" void kernel_launch(void* const* d_in, const int* in_sizes, int n_in,
                              void* d_out, int out_size, void* d_ws, size_t ws_size,
                              hipStream_t stream) {
  const float* x    = (const float*)d_in[0];   // (8,2048,1024) f32
  const float* Win  = (const float*)d_in[1];   // (4096,1024)   f32
  const float* Wout = (const float*)d_in[2];   // (4096,1024)   f32
  float* out = (float*)d_out;                  // (16384,1024)  f32

  char* ws = (char*)d_ws;
  unsigned short* xb    = (unsigned short*)ws;                               // 32 MB
  unsigned short* Winb  = xb   + (size_t)BROWS * INPUT_DIM;                  //  8 MB
  unsigned short* WoutT = Winb + (size_t)HIDDEN_DIM * INPUT_DIM;             //  2 MB
  unsigned short* topc  = WoutT + (size_t)OUTPUT_DIM * SPARSITY;             // 32 MB
  float* partial = (float*)(topc + (size_t)BROWS * SPARSITY);                //  2 MB
  float* norm    = partial + (size_t)BROWS * (HIDDEN_DIM / BN);              // 64 KB

  const int n4x = BROWS * INPUT_DIM / 4;
  convert_f32_bf16<<<(n4x + 255) / 256, 256, 0, stream>>>(x, xb, n4x);
  const int n4w = HIDDEN_DIM * INPUT_DIM / 4;
  convert_f32_bf16<<<(n4w + 255) / 256, 256, 0, stream>>>(Win, Winb, n4w);
  transpose_convert_wout<<<dim3(SPARSITY / 64, OUTPUT_DIM / 64), 256, 0, stream>>>(Wout, WoutT);

  dim3 g1(BROWS / BM, HIDDEN_DIM / BN);        // (128, 32)
  sparse_mlp_gemm1_act<<<g1, 256, 0, stream>>>(xb, Winb, topc, partial);

  sparse_mlp_rownorm<<<(BROWS + 255) / 256, 256, 0, stream>>>(partial, norm, HIDDEN_DIM / BN);

  dim3 g2(BROWS / BM, OUTPUT_DIM / BN);        // (128, 8)
  sparse_mlp_gemm2<<<g2, 256, 0, stream>>>(topc, norm, WoutT, out);
}
